// Top2Router_80685255623334
// MI455X (gfx1250) — matrix operand order repaired and checked by reading, not compiled, not run on hardware
//
#include <hip/hip_runtime.h>
#include <stdint.h>

// Problem constants (match reference)
#define B_    8
#define T_    8192
#define C_    256
#define E_    64
#define N_    (B_*T_)      // 65536 tokens
#define S_    (N_*2)       // 131072 assignments (top-2)
#define CAP_  2560         // int(1.25 * S / E)
#define RBLK  2048         // assignments per rank block
#define NRBLK (S_/RBLK)    // 64

typedef __attribute__((ext_vector_type(16))) __bf16 v16bf;
typedef __attribute__((ext_vector_type(8)))  float  v8f;

#if defined(__has_builtin)
#  if __has_builtin(__builtin_amdgcn_tensor_load_to_lds)
#    define HAVE_TDM 1
#  endif
#endif
#ifndef HAVE_TDM
#  define HAVE_TDM 0
#endif

static __device__ __forceinline__ unsigned short f2bf(float f) {
  unsigned u = __float_as_uint(f);
  unsigned r = u + 0x7FFFu + ((u >> 16) & 1u);   // round-to-nearest-even
  return (unsigned short)(r >> 16);
}

// ---------------- init kernels ----------------
__global__ void k_zero(float* __restrict__ p, int n) {
  int i = (blockIdx.x * blockDim.x + threadIdx.x) * 4;
  if (i + 4 <= n) {
    float4 z = {0.f, 0.f, 0.f, 0.f};
    *(float4*)(p + i) = z;
  } else {
    for (int j = i; j < n; ++j) p[j] = 0.f;
  }
}

__global__ void k_init_slots(int* __restrict__ slot_tok) {
  int i = blockIdx.x * 256 + threadIdx.x;
  if (i < E_ * CAP_) slot_tok[i] = -1;
}

// Convert W_experts f32 [E][K=C][N=C] -> bf16 TRANSPOSED [E][N][K] so that the
// GEMM B-tile staging is a plain row-major block copy (TDM friendly) and
// B fragments are single ds_load_b32 per VGPR (K-pairs contiguous).
__global__ void k_convWT(const float* __restrict__ W, unsigned short* __restrict__ Wt) {
  __shared__ float t[32][33];
  const int e  = blockIdx.z;
  const int k0 = blockIdx.y * 32;
  const int n0 = blockIdx.x * 32;
  const int c  = threadIdx.x & 31, r4 = threadIdx.x >> 5;
  const float* src = W + (size_t)e * C_ * C_;
  unsigned short* dst = Wt + (size_t)e * C_ * C_;
  for (int i = 0; i < 4; ++i) {
    int r = r4 * 4 + i;
    t[r][c] = src[(size_t)(k0 + r) * C_ + n0 + c];
  }
  __syncthreads();
  for (int i = 0; i < 4; ++i) {
    int r = r4 * 4 + i;
    dst[(size_t)(n0 + r) * C_ + k0 + c] = f2bf(t[c][r]);
  }
}

// ---------------- gating: logits, softmax, top-2, gate partial sums ----------------
__global__ void k_gate(const float* __restrict__ x, const float* __restrict__ Wg,
                       int* __restrict__ eid, float* __restrict__ sc,
                       float* __restrict__ gateP) {
  __shared__ float xs[8][C_];
  __shared__ float gb[8][E_];
  const int tid = threadIdx.x, wave = tid >> 5, lane = tid & 31;
  const int tokBase = blockIdx.x * 8;

  for (int i = 0; i < 8; ++i) {
    int idx = tid + i * 256;
    xs[idx >> 8][idx & 255] = x[(size_t)(tokBase + (idx >> 8)) * C_ + (idx & 255)];
  }
  __syncthreads();

  const int tok = tokBase + wave;
  float a0 = 0.f, a1 = 0.f;                    // experts lane, lane+32
  for (int c = 0; c < C_; ++c) {
    float xv = xs[wave][c];
    a0 = fmaf(xv, Wg[c * E_ + lane], a0);
    a1 = fmaf(xv, Wg[c * E_ + lane + 32], a1);
  }
  float m = fmaxf(a0, a1);
  for (int o = 16; o > 0; o >>= 1) m = fmaxf(m, __shfl_xor(m, o, 32));
  float e0 = __expf(a0 - m), e1 = __expf(a1 - m);
  float sum = e0 + e1;
  for (int o = 16; o > 0; o >>= 1) sum += __shfl_xor(sum, o, 32);
  float g0 = e0 / sum, g1 = e1 / sum;
  gb[wave][lane] = g0; gb[wave][lane + 32] = g1;

  float bv = (g0 >= g1) ? g0 : g1; int be = (g0 >= g1) ? lane : lane + 32;
  for (int o = 16; o > 0; o >>= 1) {
    float ov = __shfl_xor(bv, o, 32); int oe = __shfl_xor(be, o, 32);
    if (ov > bv || (ov == bv && oe < be)) { bv = ov; be = oe; }
  }
  float h0 = (lane == be) ? -1.f : g0;
  float h1 = (lane + 32 == be) ? -1.f : g1;
  float cv = (h0 >= h1) ? h0 : h1; int ce = (h0 >= h1) ? lane : lane + 32;
  for (int o = 16; o > 0; o >>= 1) {
    float ov = __shfl_xor(cv, o, 32); int oe = __shfl_xor(ce, o, 32);
    if (ov > cv || (ov == cv && oe < ce)) { cv = ov; ce = oe; }
  }
  if (lane == 0) {
    eid[tok * 2]     = be; sc[tok * 2]     = bv;
    eid[tok * 2 + 1] = ce; sc[tok * 2 + 1] = cv;
  }
  __syncthreads();
  if (tid < E_) {
    float acc = 0.f;
    for (int t = 0; t < 8; ++t) acc += gb[t][tid];
    gateP[blockIdx.x * E_ + tid] = acc;
  }
}

// ---------------- FIFO rank scan (exact, deterministic) ----------------
__global__ void k_count(const int* __restrict__ eid, int* __restrict__ blkCnt) {
  __shared__ int cnt[E_];
  int tid = threadIdx.x;
  if (tid < E_) cnt[tid] = 0;
  __syncthreads();
  int base = blockIdx.x * RBLK;
  for (int i = 0; i < RBLK / 256; ++i)
    atomicAdd(&cnt[eid[base + tid + i * 256]], 1);
  __syncthreads();
  if (tid < E_) blkCnt[blockIdx.x * E_ + tid] = cnt[tid];
}

__global__ void k_offset(const int* __restrict__ blkCnt, int* __restrict__ blkOff) {
  int e = threadIdx.x;
  int run = 0;
  for (int b = 0; b < NRBLK; ++b) {
    blkOff[b * E_ + e] = run;
    run += blkCnt[b * E_ + e];
  }
}

__global__ void k_rank(const int* __restrict__ eid, const float* __restrict__ sc,
                       int* __restrict__ bidx, int* __restrict__ slot_tok,
                       float* __restrict__ slot_sc, const int* __restrict__ blkOff) {
  __shared__ int cnt[E_];
  const int lane = threadIdx.x;                // single wave32
  cnt[lane]      = blkOff[blockIdx.x * E_ + lane];
  cnt[lane + 32] = blkOff[blockIdx.x * E_ + lane + 32];
  __syncthreads();
  const int base = blockIdx.x * RBLK;
  for (int g = 0; g < RBLK / 32; ++g) {
    int s = base + g * 32 + lane;
    int e = eid[s];
    int prior = cnt[e];
    int before = 0;
    for (int j = 0; j < 32; ++j) {
      int oe = __shfl(e, j, 32);
      if (j < lane && oe == e) before++;
    }
    int rank = prior + before;
    int bi = (rank < CAP_) ? (e * CAP_ + rank) : -1;
    bidx[s] = bi;
    if (bi >= 0) { slot_tok[bi] = s >> 1; slot_sc[bi] = sc[s]; }
    atomicAdd(&cnt[e], 1);
    __syncthreads();
  }
}

// ---------------- scatter kept token rows into per-expert bf16 buffers ----------------
__global__ void k_scatter(const float* __restrict__ x, const int* __restrict__ bidx,
                          unsigned short* __restrict__ buf) {
  int s = blockIdx.x;
  int bi = bidx[s];
  if (bi < 0) return;
  int tok = s >> 1;
  int c = threadIdx.x * 4;
  const float4 v = *(const float4*)(x + (size_t)tok * C_ + c);
  unsigned lo = (unsigned)f2bf(v.x) | ((unsigned)f2bf(v.y) << 16);
  unsigned hi = (unsigned)f2bf(v.z) | ((unsigned)f2bf(v.w) << 16);
  unsigned* dst = (unsigned*)(buf + (size_t)bi * C_ + c);
  dst[0] = lo; dst[1] = hi;
}

// ---------------- TDM 2-D tile load: global(bf16, row stride 256 elems) -> LDS ----------------
#if HAVE_TDM
typedef __attribute__((ext_vector_type(4))) unsigned int v4u;
typedef __attribute__((ext_vector_type(8))) int v8i;
typedef __attribute__((ext_vector_type(4))) int v4i;

static __device__ __forceinline__ void tdm_load_2d(unsigned ldsByteAddr, const void* gp,
                                                   unsigned tileW, unsigned tileH,
                                                   unsigned rowStrideElems) {
  unsigned long long ga = (unsigned long long)(size_t)gp;
  // D# group0: count=1, lds_addr, global_addr[56:0], type=2 ("image")
  v4u g0;
  g0[0] = 1u;
  g0[1] = ldsByteAddr;
  g0[2] = (unsigned)(ga & 0xFFFFFFFFu);
  g0[3] = (unsigned)((ga >> 32) & 0x01FFFFFFu) | (2u << 30);
  // D# group1: data_size=1(2B), pad_enable, pad_interval=3(16 dwords), pad_amount=0(1 dword)
  // tensor dims set large (1<<20) so no OOB clamping; tile = tileW x tileH; dim0 stride.
  const unsigned DIM = 1u << 20;
  v8i g1;
  g1[0] = (int)((1u << 16) | (1u << 20) | (3u << 22));
  g1[1] = (int)((DIM & 0xFFFFu) << 16);                       // tensor_dim0 lo16
  g1[2] = (int)((DIM >> 16) | ((DIM & 0xFFFFu) << 16));       // dim0 hi16 | dim1 lo16
  g1[3] = (int)((DIM >> 16) | (tileW << 16));                 // dim1 hi16 | tile_dim0
  g1[4] = (int)(tileH & 0xFFFFu);                             // tile_dim1 (tile_dim2=0)
  g1[5] = (int)rowStrideElems;                                // tensor_dim0_stride lo32
  g1[6] = 0;
  g1[7] = 0;
  v4i z4 = {0, 0, 0, 0};
  v8i z8 = {0, 0, 0, 0, 0, 0, 0, 0};
  __builtin_amdgcn_tensor_load_to_lds(g0, g1, z4, z4, z8, 0);
}
#endif

// ---------------- grouped GEMM: WMMA bf16->f32, fused scale + scatter-add ----------------
#define MT 128
#define NT 128
#define KT 32
#define STGDW (128*17)   // 128 rows x (16 data + 1 pad) dwords

// Fragment row base pointers are precomputed (loop-invariant); all inner LDS
// loads use constant immediate offsets: A = {0..7}, B = {0..7}.
static __device__ __forceinline__ void mma_step(const unsigned* const (&aR)[2],
                                                const unsigned* const (&bR)[4],
                                                v8f (&acc)[2][4]) {
  union Frag { unsigned u[8]; v16bf v; };
  Frag a[2], b[4];
  #pragma unroll
  for (int tm = 0; tm < 2; ++tm) {
    a[tm].u[0] = aR[tm][0];  a[tm].u[1] = aR[tm][1];
    a[tm].u[2] = aR[tm][2];  a[tm].u[3] = aR[tm][3];
    a[tm].u[4] = aR[tm][4];  a[tm].u[5] = aR[tm][5];
    a[tm].u[6] = aR[tm][6]; a[tm].u[7] = aR[tm][7];
  }
  #pragma unroll
  for (int tn = 0; tn < 4; ++tn) {
    #pragma unroll
    for (int v = 0; v < 8; ++v) b[tn].u[v] = bR[tn][v];
  }
  #pragma unroll
  for (int tm = 0; tm < 2; ++tm)
    #pragma unroll
    for (int tn = 0; tn < 4; ++tn)
      acc[tm][tn] = __builtin_amdgcn_wmma_f32_16x16x32_bf16(
          false, a[tm].v, false, b[tn].v, (short)0, acc[tm][tn], false, false);
}

__global__ __launch_bounds__(256) void k_gemm(const unsigned short* __restrict__ bufb,
                       const unsigned short* __restrict__ Wbt,
                       const int* __restrict__ slot_tok,
                       const float* __restrict__ slot_sc,
                       float* __restrict__ out) {
#if HAVE_TDM
  __shared__ unsigned stg[2][2][STGDW];        // [buf][A/B], TDM double-buffered
#else
  __shared__ unsigned stg[1][2][STGDW];
#endif
  const int tid = threadIdx.x;
  const int lane = tid & 31, wid = tid >> 5;
  const int half = lane >> 4, l15 = lane & 15;
  const int waveM = wid & 3, waveN = wid >> 2; // 4x2 wave grid, wave tile 32x64
  const int e = blockIdx.z;
  const int tileM = blockIdx.y * MT;
  const int tileN = blockIdx.x * NT;
  const unsigned short* aBase = bufb + (size_t)e * CAP_ * C_ + (size_t)tileM * C_;
  const unsigned short* bBase = Wbt  + (size_t)e * C_  * C_ + (size_t)tileN * C_;

  // Loop-invariant fragment row bases (half-offset folded in):
  // A frag (16x32 bf16): lanes 0-15 K 0..15 ; lanes 16-31 K 16..31
  // B frag (32x16 bf16): lanes 0-15 K 0..15 ; lanes 16-31 K 16..31
  const unsigned *aR0[2], *bR0[4];
#if HAVE_TDM
  const unsigned *aR1[2], *bR1[4];
#endif
  for (int tm = 0; tm < 2; ++tm) {
    int row = waveM * 32 + tm * 16 + l15;
    aR0[tm] = &stg[0][0][row * 17 + half * 8];
#if HAVE_TDM
    aR1[tm] = &stg[1][0][row * 17 + half * 8];
#endif
  }
  for (int tn = 0; tn < 4; ++tn) {
    int n = waveN * 64 + tn * 16 + l15;
    bR0[tn] = &stg[0][1][n * 17 + half * 8];
#if HAVE_TDM
    bR1[tn] = &stg[1][1][n * 17 + half * 8];
#endif
  }

  v8f acc[2][4] = {};

#if HAVE_TDM
  if (wid == 0) {                              // wave 0 drives the DMA engine
    tdm_load_2d((unsigned)(size_t)(void*)&stg[0][0][0], aBase, KT, MT, C_);
    tdm_load_2d((unsigned)(size_t)(void*)&stg[0][1][0], bBase, KT, NT, C_);
  }
  // Software pipeline, unrolled x2 so buffer selection is static (no spills).
  for (int k0 = 0; k0 < C_; k0 += 2 * KT) {
    // ---- phase 0: prefetch buf1 @ k0+KT, compute buf0 @ k0 ----
    __syncthreads();                           // buf1 free for reuse
    if (wid == 0) {
      tdm_load_2d((unsigned)(size_t)(void*)&stg[1][0][0], aBase + (k0 + KT), KT, MT, C_);
      tdm_load_2d((unsigned)(size_t)(void*)&stg[1][1][0], bBase + (k0 + KT), KT, NT, C_);
      __builtin_amdgcn_s_wait_tensorcnt(2);    // buf0 pair retired
    }
    __syncthreads();                           // publish buf0
    mma_step(aR0, bR0, acc);
    // ---- phase 1: prefetch buf0 @ k0+2KT, compute buf1 @ k0+KT ----
    __syncthreads();                           // buf0 free for reuse
    if (wid == 0) {
      if (k0 + 2 * KT < C_) {
        tdm_load_2d((unsigned)(size_t)(void*)&stg[0][0][0], aBase + (k0 + 2 * KT), KT, MT, C_);
        tdm_load_2d((unsigned)(size_t)(void*)&stg[0][1][0], bBase + (k0 + 2 * KT), KT, NT, C_);
        __builtin_amdgcn_s_wait_tensorcnt(2);  // buf1 pair retired
      } else {
        __builtin_amdgcn_s_wait_tensorcnt(0);
      }
    }
    __syncthreads();                           // publish buf1
    mma_step(aR1, bR1, acc);
  }
#else
  const unsigned* aDw = (const unsigned*)aBase;  // row stride 128 dwords
  const unsigned* bDw = (const unsigned*)bBase;
  for (int k0 = 0; k0 < C_; k0 += KT) {
    __syncthreads();
    for (int i = 0; i < 8; ++i) {              // A: 128 rows x 16 dwords
      int idx = tid + i * 256;
      int r = idx >> 4, c2 = idx & 15;
      stg[0][0][r * 17 + c2] = aDw[(size_t)r * (C_ / 2) + (k0 >> 1) + c2];
    }
    for (int i = 0; i < 8; ++i) {              // B (pre-transposed): 128 rows x 16 dwords
      int idx = tid + i * 256;
      int r = idx >> 4, c2 = idx & 15;
      stg[0][1][r * 17 + c2] = bDw[(size_t)r * (C_ / 2) + (k0 >> 1) + c2];
    }
    __syncthreads();
    mma_step(aR0, bR0, acc);
  }
#endif

  // epilogue: scale by gate score, scatter-add into token rows.
  // Each token receives at most 2 f32 contributions; addition is commutative -> deterministic.
  for (int tm = 0; tm < 2; ++tm) {
    #pragma unroll
    for (int i = 0; i < 8; ++i) {
      int slot = tileM + waveM * 32 + tm * 16 + i + half * 8;
      int gslot = e * CAP_ + slot;
      int tok = slot_tok[gslot];
      if (tok < 0) continue;
      float scv = slot_sc[gslot];
      #pragma unroll
      for (int tn = 0; tn < 4; ++tn) {
        int n = tileN + waveN * 64 + tn * 16 + l15;
        atomicAdd(out + (size_t)tok * C_ + n, acc[tm][tn][i] * scv);
      }
    }
  }
}

// ---------------- aux loss ----------------
__global__ void k_aux(const float* __restrict__ gateP, float* __restrict__ out) {
  __shared__ float sq[E_];
  int e = threadIdx.x;
  float s = 0.f;
  for (int b = 0; b < N_ / 8; ++b) s += gateP[b * E_ + e];
  float load = s / (float)N_;
  float d = load - 1.0f / (float)E_;
  sq[e] = d * d;
  __syncthreads();
  if (e == 0) {
    float t = 0.f;
    for (int i = 0; i < E_; ++i) t += sq[i];
    out[(size_t)N_ * C_] = t / (float)E_;
  }
}

extern "C" void kernel_launch(void* const* d_in, const int* in_sizes, int n_in,
                              void* d_out, int out_size, void* d_ws, size_t ws_size,
                              hipStream_t stream) {
  (void)in_sizes; (void)n_in; (void)out_size; (void)ws_size;
  const float* x  = (const float*)d_in[0];   // [B,T,C]
  const float* Wg = (const float*)d_in[1];   // [C,E]
  const float* We = (const float*)d_in[2];   // [E,C,C]
  float* out = (float*)d_out;                // [B,T,C] + aux scalar

  char* ws = (char*)d_ws;
  unsigned short* Wbt  = (unsigned short*)ws; ws += (size_t)E_ * C_ * C_ * 2;    // 8 MB (transposed)
  unsigned short* bufb = (unsigned short*)ws; ws += (size_t)E_ * CAP_ * C_ * 2;  // 84 MB
  int*   eid      = (int*)ws;   ws += (size_t)S_ * 4;
  float* sc       = (float*)ws; ws += (size_t)S_ * 4;
  int*   bidx     = (int*)ws;   ws += (size_t)S_ * 4;
  int*   slot_tok = (int*)ws;   ws += (size_t)E_ * CAP_ * 4;
  float* slot_sc  = (float*)ws; ws += (size_t)E_ * CAP_ * 4;
  int*   blkCnt   = (int*)ws;   ws += (size_t)NRBLK * E_ * 4;
  int*   blkOff   = (int*)ws;   ws += (size_t)NRBLK * E_ * 4;
  float* gateP    = (float*)ws; ws += (size_t)(N_ / 8) * E_ * 4;

  const int outN = N_ * C_ + 1;
  k_zero<<<(outN + 1023) / 1024, 256, 0, stream>>>(out, outN);
  k_init_slots<<<(E_ * CAP_ + 255) / 256, 256, 0, stream>>>(slot_tok);
  k_convWT<<<dim3(C_ / 32, C_ / 32, E_), 256, 0, stream>>>(We, Wbt);
  k_gate<<<N_ / 8, 256, 0, stream>>>(x, Wg, eid, sc, gateP);
  k_count<<<NRBLK, 256, 0, stream>>>(eid, blkCnt);
  k_offset<<<1, 64, 0, stream>>>(blkCnt, blkOff);
  k_rank<<<NRBLK, 32, 0, stream>>>(eid, sc, bidx, slot_tok, slot_sc, blkOff);
  k_scatter<<<S_, 64, 0, stream>>>(x, bidx, bufb);
  k_gemm<<<dim3(C_ / NT, CAP_ / MT, E_), 256, 0, stream>>>(bufb, Wbt, slot_tok, slot_sc, out);
  k_aux<<<1, 64, 0, stream>>>(gateP, out);
}
